// MultiHeadAttentionBlock_16681652978365
// MI455X (gfx1250) — matrix operand order repaired and checked by reading, not compiled, not run on hardware
//
#include <hip/hip_runtime.h>
#include <math.h>

// Problem constants (match reference)
constexpr int BATCH = 4;
constexpr int SEQ   = 2048;
constexpr int DIM   = 1024;
constexpr int NH    = 16;
constexpr int DKC   = 64;           // head dim
constexpr int MROWS = BATCH * SEQ;  // 8192

typedef __attribute__((ext_vector_type(16))) _Float16 v16h;
typedef __attribute__((ext_vector_type(8)))  _Float16 v8h;
typedef __attribute__((ext_vector_type(8)))  float    v8f;
typedef __attribute__((ext_vector_type(4)))  float    v4f;

// ---------------------------------------------------------------------------
// f16 fragment loaders (wave32 WMMA layouts, CDNA5 ISA §7.12.2).
// A 16x32: lanes 0-15 row M=lane, K {kb..kb+7, kb+16..23};
//          lanes 16-31 row M=lane-16, K {kb+8..15, kb+24..31}.
// B 32x16: lane holds column N=lane&15; lanes 0-15 K {kb..kb+7, kb+16..23},
//          lanes 16-31 K {kb+8..15, kb+24..31} (same K order as A).
// ---------------------------------------------------------------------------
__device__ inline v16h load_a_frag_h(const _Float16* __restrict__ base, int ld,
                                     int row0, int kb, int lane) {
  const _Float16* p = base + (size_t)(row0 + (lane & 15)) * ld + kb + ((lane & 16) ? 8 : 0);
  const v8h c0 = *(const v8h*)p;
  const v8h c1 = *(const v8h*)(p + 16);
  v16h a;
#pragma unroll
  for (int i = 0; i < 8; ++i) { a[i] = c0[i]; a[8 + i] = c1[i]; }
  return a;
}

__device__ inline v16h load_b_frag_h(const _Float16* __restrict__ base, int ld,
                                     int ncol, int kb, int lane) {
  const _Float16* p = base + (size_t)(ncol + (lane & 15)) * ld + kb + ((lane & 16) ? 8 : 0);
  const v8h c0 = *(const v8h*)p;
  const v8h c1 = *(const v8h*)(p + 16);
  v16h b;
#pragma unroll
  for (int i = 0; i < 8; ++i) { b[i] = c0[i]; b[8 + i] = c1[i]; }
  return b;
}

__device__ inline v8f wmma16(v16h a, v16h b, v8f c) {
  return __builtin_amdgcn_wmma_f32_16x16x32_f16(false, a, false, b,
                                                (short)0, c, false, false);
}

// Row reductions inside one 16-lane half (XOR masks 1..8 stay in the half).
__device__ inline float half_max(float x) {
#pragma unroll
  for (int off = 8; off >= 1; off >>= 1) x = fmaxf(x, __shfl_xor(x, off, 32));
  return x;
}
__device__ inline float half_sum(float x) {
#pragma unroll
  for (int off = 8; off >= 1; off >>= 1) x += __shfl_xor(x, off, 32);
  return x;
}

// ---------------------------------------------------------------------------
// One-time fp32 -> f16 conversion (vectorized, 8 elems/thread).
// ---------------------------------------------------------------------------
__global__ __launch_bounds__(256) void cvt_f16_kernel(const float* __restrict__ src,
                                                      _Float16* __restrict__ dst,
                                                      int n8) {
  const int i = blockIdx.x * blockDim.x + threadIdx.x;
  if (i >= n8) return;
  const v4f* s = (const v4f*)(src + (size_t)i * 8);
  const v4f x0 = s[0], x1 = s[1];
  v8h o;
#pragma unroll
  for (int j = 0; j < 4; ++j) { o[j] = (_Float16)x0[j]; o[4 + j] = (_Float16)x1[j]; }
  *(v8h*)(dst + (size_t)i * 8) = o;
}

// ---------------------------------------------------------------------------
// Projection GEMM (f16 in, fp32 accumulate): Y[m,n] = sum_k X[m,k]*W[n,k]
//   mode 0: Y -> outh  [B,H,S,DK] f16                       (Q-proj)
//   mode 1: Y -> out0  [B,H,S,DK] fp32  + outh f16 same     (K-proj)
//   mode 2: Y -> out0  [B,H,S,DK] fp32  + outh f16 [B,H,DK,S] transposed (V)
//   mode 3: Y -> out0  row-major [M,N] fp32                 (O-proj)
// 256 thr = 8 waves as 4(M) x 2(N) of 32x64 wave tiles -> 128x128 block tile.
// f16 working set is L2-resident (192 MB) so fragments load straight from L2.
// ---------------------------------------------------------------------------
__global__ __launch_bounds__(256) void proj_gemm_kernel(
    const _Float16* __restrict__ X, const _Float16* __restrict__ W,
    float* __restrict__ out0, _Float16* __restrict__ outh, int mode) {
  const int lane = threadIdx.x & 31;
  const int wv   = threadIdx.x >> 5;
  const int m0   = blockIdx.y * 128 + (wv >> 1) * 32;
  const int n0   = blockIdx.x * 128 + (wv & 1) * 64;

  v8f zero = {};
  v8f acc[2][4];
#pragma unroll
  for (int mi = 0; mi < 2; ++mi)
#pragma unroll
    for (int ni = 0; ni < 4; ++ni) acc[mi][ni] = zero;

  for (int kb = 0; kb < DIM; kb += 32) {
    const v16h a0 = load_a_frag_h(X, DIM, m0, kb, lane);
    const v16h a1 = load_a_frag_h(X, DIM, m0 + 16, kb, lane);
#pragma unroll
    for (int ni = 0; ni < 4; ++ni) {
      const v16h b = load_b_frag_h(W, DIM, n0 + ni * 16, kb, lane);
      acc[0][ni] = wmma16(a0, b, acc[0][ni]);
      acc[1][ni] = wmma16(a1, b, acc[1][ni]);
    }
  }

  // C layout: lanes 0-15: VGPR v -> (M=v, N=lane); lanes 16-31: (M=v+8, N=lane-16)
  const int rbase = (lane & 16) ? 8 : 0;
  const int cl    = lane & 15;
#pragma unroll
  for (int mi = 0; mi < 2; ++mi) {
#pragma unroll
    for (int ni = 0; ni < 4; ++ni) {
#pragma unroll
      for (int vv = 0; vv < 8; ++vv) {
        const int   row = m0 + mi * 16 + rbase + vv;
        const int   col = n0 + ni * 16 + cl;
        const float val = acc[mi][ni][vv];
        if (mode == 3) {
          out0[(size_t)row * DIM + col] = val;
        } else {
          const int    bb  = row >> 11, ss = row & (SEQ - 1);
          const int    hh  = col >> 6,  dk = col & (DKC - 1);
          const size_t idx = (((size_t)(bb * NH + hh)) * SEQ + ss) * DKC + dk;
          if (mode == 0) {
            outh[idx] = (_Float16)val;
          } else if (mode == 1) {
            out0[idx] = val;
            outh[idx] = (_Float16)val;
          } else {  // mode 2
            out0[idx] = val;
            outh[(((size_t)(bb * NH + hh)) * DKC + dk) * SEQ + ss] = (_Float16)val;
          }
        }
      }
    }
  }
}

// ---------------------------------------------------------------------------
// Causal flash attention, all-f16 operands, fp32 accumulate.
// Q,K in [B,H,S,DK] f16; V transposed [B,H,DK,S] f16; ctx out [B,S,D] f16.
// 8 waves/block; each wave owns 32 query rows (2 row-fragments kept resident),
// streams causal key tiles of 32. Each K/V fragment feeds 2 WMMAs.
// ---------------------------------------------------------------------------
__global__ __launch_bounds__(256) void flash_attn_kernel(
    const _Float16* __restrict__ Q, const _Float16* __restrict__ K,
    const _Float16* __restrict__ Vt, _Float16* __restrict__ Ctx) {
  __shared__ _Float16 ldsP[8][32][32];  // per-wave 32x32 f16 P tile (16 KB)

  const int lane = threadIdx.x & 31;
  const int wv   = threadIdx.x >> 5;
  const int bh   = blockIdx.y;
  const int b    = bh >> 4;
  const int h    = bh & 15;
  const int q0   = blockIdx.x * 256 + wv * 32;  // query row base within sequence

  const _Float16* Qh = Q + (size_t)bh * SEQ * DKC;
  const _Float16* Kh = K + (size_t)bh * SEQ * DKC;
  const _Float16* Vh = Vt + (size_t)bh * DKC * SEQ;

  // Resident Q fragments: [row block][d chunk]
  v16h aq[2][2];
#pragma unroll
  for (int qi = 0; qi < 2; ++qi) {
    aq[qi][0] = load_a_frag_h(Qh, DKC, q0 + qi * 16, 0, lane);
    aq[qi][1] = load_a_frag_h(Qh, DKC, q0 + qi * 16, 32, lane);
  }

  v8f zero = {};
  v8f co[2][4];
  float mrow[2][8], lrow[2][8];
#pragma unroll
  for (int qi = 0; qi < 2; ++qi) {
#pragma unroll
    for (int j = 0; j < 4; ++j) co[qi][j] = zero;
#pragma unroll
    for (int i = 0; i < 8; ++i) { mrow[qi][i] = -3.0e38f; lrow[qi][i] = 0.0f; }
  }

  const int   rbase = (lane & 16) ? 8 : 0;
  const int   cl    = lane & 15;
  const float scale = 0.125f;  // 1/sqrt(64)

  // Causal: this wave needs keys k < q0+32.
  for (int kb = 0; kb < q0 + 32; kb += 32) {
    // prefetch next key/value tiles into cache (speculative, gfx1250 path)
    if (kb + 32 < q0 + 32) {
      __builtin_prefetch(Kh + (size_t)(kb + 32 + cl) * DKC, 0, 0);
      __builtin_prefetch(Vh + (size_t)cl * SEQ + kb + 32, 0, 0);
    }

    // K fragments: [n chunk][d chunk]
    v16h bk[2][2];
#pragma unroll
    for (int n = 0; n < 2; ++n) {
      bk[n][0] = load_b_frag_h(Kh, DKC, kb + n * 16, 0, lane);
      bk[n][1] = load_b_frag_h(Kh, DKC, kb + n * 16, 32, lane);
    }

#pragma unroll
    for (int qi = 0; qi < 2; ++qi) {
      v8f s0 = zero, s1 = zero;
      s0 = wmma16(aq[qi][0], bk[0][0], s0);
      s0 = wmma16(aq[qi][1], bk[0][1], s0);
      s1 = wmma16(aq[qi][0], bk[1][0], s1);
      s1 = wmma16(aq[qi][1], bk[1][1], s1);

      // online softmax for this 16-row block; P -> LDS f16
#pragma unroll
      for (int vv = 0; vv < 8; ++vv) {
        const int row = q0 + qi * 16 + rbase + vv;
        const int c0  = kb + cl;
        const int c1  = kb + 16 + cl;
        float x0 = s0[vv] * scale; if (c0 > row) x0 = -1.0e30f;
        float x1 = s1[vv] * scale; if (c1 > row) x1 = -1.0e30f;
        const float tmax  = half_max(fmaxf(x0, x1));
        const float mnew  = fmaxf(mrow[qi][vv], tmax);
        const float alpha = __expf(mrow[qi][vv] - mnew);
        const float p0    = __expf(x0 - mnew);
        const float p1    = __expf(x1 - mnew);
        lrow[qi][vv] = lrow[qi][vv] * alpha + half_sum(p0 + p1);
        mrow[qi][vv] = mnew;
#pragma unroll
        for (int j = 0; j < 4; ++j) co[qi][j][vv] *= alpha;
        ldsP[wv][qi * 16 + rbase + vv][cl]      = (_Float16)p0;
        ldsP[wv][qi * 16 + rbase + vv][cl + 16] = (_Float16)p1;
      }
    }

    // C-layout -> A-layout transform through per-wave LDS tile; split DS
    // counter wait (CDNA5) keeps the stores visible to the reload.
    asm volatile("s_wait_dscnt 0x0" ::: "memory");
    v16h pa[2];
#pragma unroll
    for (int qi = 0; qi < 2; ++qi) {
      const _Float16* prow = &ldsP[wv][qi * 16 + cl][0];
      const int off = (lane & 16) ? 8 : 0;
#pragma unroll
      for (int i = 0; i < 8; ++i) {
        pa[qi][i]     = prow[off + i];
        pa[qi][8 + i] = prow[off + 16 + i];
      }
    }

    // ctx(32x64) += P(32x32) x V(32x64): each V fragment feeds both row blocks
#pragma unroll
    for (int j = 0; j < 4; ++j) {
      const v16h bv = load_b_frag_h(Vh, SEQ, j * 16, kb, lane);
      co[0][j] = wmma16(pa[0], bv, co[0][j]);
      co[1][j] = wmma16(pa[1], bv, co[1][j]);
    }
  }

  // normalize and scatter ctx (f16) into [B,S,D]
#pragma unroll
  for (int qi = 0; qi < 2; ++qi) {
#pragma unroll
    for (int j = 0; j < 4; ++j) {
#pragma unroll
      for (int vv = 0; vv < 8; ++vv) {
        const int s = q0 + qi * 16 + rbase + vv;
        const int d = j * 16 + cl;
        Ctx[((size_t)b * SEQ + s) * DIM + h * DKC + d] =
            (_Float16)(co[qi][j][vv] / lrow[qi][vv]);
      }
    }
  }
}

// ---------------------------------------------------------------------------
// Launch: fp32->f16 once -> Q/K/V projections -> flash attention -> O-proj.
// d_out = [ out (B*S*D) | key_h (B*H*S*DK) | val_h (B*H*S*DK) ]  fp32
// d_ws (f16): Xq,Xk,Xv (3x16MB) | Wq,Wk,Wv,Wo (4x2MB) | Q,K (2x16MB)
//             | V^T (16MB) | ctx (16MB)  = 120 MB
// ---------------------------------------------------------------------------
extern "C" void kernel_launch(void* const* d_in, const int* in_sizes, int n_in,
                              void* d_out, int out_size, void* d_ws, size_t ws_size,
                              hipStream_t stream) {
  (void)in_sizes; (void)n_in; (void)out_size; (void)ws_size;
  const float* q   = (const float*)d_in[0];
  const float* k   = (const float*)d_in[1];
  const float* v   = (const float*)d_in[2];
  // d_in[3] is the causal mask (int32 tril) — structure is known statically.
  const float* w_q = (const float*)d_in[4];
  const float* w_k = (const float*)d_in[5];
  const float* w_v = (const float*)d_in[6];
  const float* w_o = (const float*)d_in[7];

  const size_t NE = (size_t)BATCH * SEQ * DIM;  // 8,388,608
  const size_t NW = (size_t)DIM * DIM;          // 1,048,576

  float* out     = (float*)d_out;
  float* key_out = out + NE;
  float* val_out = out + 2 * NE;

  _Float16* wsh  = (_Float16*)d_ws;
  _Float16* Xq   = wsh;                 // [B,S,D] f16
  _Float16* Xk   = Xq + NE;
  _Float16* Xv   = Xk + NE;
  _Float16* Wq   = Xv + NE;             // [D,D] f16 x4
  _Float16* Wk   = Wq + NW;
  _Float16* Wv   = Wk + NW;
  _Float16* Wo   = Wv + NW;
  _Float16* Qh   = Wo + NW;             // [B,H,S,DK] f16
  _Float16* Kh   = Qh + NE;             // [B,H,S,DK] f16
  _Float16* Vth  = Kh + NE;             // [B,H,DK,S] f16
  _Float16* Ctxh = Vth + NE;            // [B,S,D]    f16

  const dim3 blk(256);
  const int  cvtE = (int)(NE / 8 / 256);  // 4096 blocks
  const int  cvtW = (int)(NW / 8 / 256);  // 512 blocks

  cvt_f16_kernel<<<cvtE, blk, 0, stream>>>(q, Xq, (int)(NE / 8));
  cvt_f16_kernel<<<cvtE, blk, 0, stream>>>(k, Xk, (int)(NE / 8));
  cvt_f16_kernel<<<cvtE, blk, 0, stream>>>(v, Xv, (int)(NE / 8));
  cvt_f16_kernel<<<cvtW, blk, 0, stream>>>(w_q, Wq, (int)(NW / 8));
  cvt_f16_kernel<<<cvtW, blk, 0, stream>>>(w_k, Wk, (int)(NW / 8));
  cvt_f16_kernel<<<cvtW, blk, 0, stream>>>(w_v, Wv, (int)(NW / 8));
  cvt_f16_kernel<<<cvtW, blk, 0, stream>>>(w_o, Wo, (int)(NW / 8));

  const dim3 gemm_grid(DIM / 128, MROWS / 128);  // (8, 64)
  proj_gemm_kernel<<<gemm_grid, blk, 0, stream>>>(Xq, Wq, nullptr, Qh, 0);
  proj_gemm_kernel<<<gemm_grid, blk, 0, stream>>>(Xk, Wk, key_out, Kh, 1);
  proj_gemm_kernel<<<gemm_grid, blk, 0, stream>>>(Xv, Wv, val_out, Vth, 2);

  flash_attn_kernel<<<dim3(SEQ / 256, BATCH * NH), blk, 0, stream>>>(Qh, Kh, Vth, Ctxh);

  proj_gemm_kernel<<<gemm_grid, blk, 0, stream>>>(Ctxh, Wo, out, nullptr, 3);
}
